// GAT_88656714925433
// MI455X (gfx1250) — matrix operand order repaired and checked by reading, not compiled, not run on hardware
//
#include <hip/hip_runtime.h>
#include <hip/hip_bf16.h>

// ---------------------------------------------------------------------------
// GAT (4x GATConv + mean-pool + FC) for MI455X / gfx1250.
//  - Dense transforms: v_wmma_f32_16x16x32_bf16, bf16 operands pre-staged
//    (A mirror + transposed weights) so the inner loop is 4x b128 loads + WMMA.
//  - Edge softmax/scatter: flat atomic passes (h = 25.6MB, L2-resident),
//    with global_prefetch warming of gather rows one pass ahead.
// ---------------------------------------------------------------------------

#define GAT_N     50000
#define GAT_E     800000
#define GAT_ETOT  (GAT_E + GAT_N)   // with self-loops
#define GAT_G     128
#define GAT_IN    128
#define GAT_HID   32
#define GAT_HEADS 4
#define GAT_OUT   64
#define GAT_SLOPE 0.2f

typedef __attribute__((ext_vector_type(16))) __bf16 v16bf;
typedef __attribute__((ext_vector_type(8)))  __bf16 v8bf;
typedef __attribute__((ext_vector_type(8)))  float  v8f;

__device__ __forceinline__ __bf16 to_bf16(float f) { return (__bf16)f; }

// Ordered-uint encoding of float for atomicMax-based segment max.
__device__ __forceinline__ unsigned int f32_ord(float f) {
  unsigned int u = __builtin_bit_cast(unsigned int, f);
  return (u & 0x80000000u) ? ~u : (u | 0x80000000u);
}
__device__ __forceinline__ float ord_f32(unsigned int u) {
  u = (u & 0x80000000u) ? (u & 0x7FFFFFFFu) : ~u;
  return __builtin_bit_cast(float, u);
}

// ---------------------------------------------------------------------------
// WMMA GEMM: C[M,Nc] = A16[M,K] @ Bt16[Nc,K]^T (+ bias[Nc]), all bf16 in,
// f32 accumulate. One wave owns a 16 x (16*NT) tile; K fully unrolled.
// Requires M%16==0, Nc % (16*NT) == 0, K%32==0.
// ---------------------------------------------------------------------------
template <int K, int NT>
__global__ void gat_wmma_gemm(const __bf16* __restrict__ A16,  // [M,K] row-major
                              const __bf16* __restrict__ Bt16, // [Nc,K] (B^T)
                              const float*  __restrict__ bias, // [Nc] or null
                              float* __restrict__ C,           // [M,Nc]
                              int M, int Nc)
{
  const int lane    = threadIdx.x & 31;
  const int wave    = threadIdx.x >> 5;
  const int wpb     = blockDim.x >> 5;
  const int tilesM  = M >> 4;
  const int groupsN = Nc / (16 * NT);
  const int tile    = blockIdx.x * wpb + wave;   // uniform per wave
  if (tile >= tilesM * groupsN) return;          // whole-wave exit, EXEC stays ~0
  const int tm = tile / groupsN;
  const int gn = tile - tm * groupsN;
  const int m0 = tm << 4;
  const int n0 = gn * (16 * NT);

  const int half = lane >> 4;   // lane group 0/1
  const int l15  = lane & 15;
  const int rowA = m0 + l15;    // both lane halves hold rows M=0..15 of the tile

  v8f acc[NT];
#pragma unroll
  for (int t = 0; t < NT; ++t) acc[t] = (v8f){};

#pragma unroll
  for (int k0 = 0; k0 < K; k0 += 32) {
    // A fragment (16-bit A 16x32 layout):
    //   half==0: K {k0..k0+7, k0+16..k0+23};  half==1: K {+8..+15, +24..+31}
    const v8bf* ap = reinterpret_cast<const v8bf*>(
        A16 + (size_t)rowA * K + k0 + half * 8);
    const v8bf alo = ap[0];   // 8 contiguous bf16  (16B load)
    const v8bf ahi = ap[2];   // +16 elements       (16B load)
    const v16bf a = __builtin_shufflevector(alo, ahi,
        0, 1, 2, 3, 4, 5, 6, 7, 8, 9, 10, 11, 12, 13, 14, 15);

#pragma unroll
    for (int t = 0; t < NT; ++t) {
      // B fragment (32x16): same K order as A: half==0: K {k0..k0+7, k0+16..k0+23};
      // half==1: K {+8..+15, +24..+31}, col = l15. Bt16 is [Nc,K].
      const v8bf* bp = reinterpret_cast<const v8bf*>(
          Bt16 + (size_t)(n0 + t * 16 + l15) * K + k0);
      const v8bf b0 = bp[half];
      const v8bf b1 = bp[half + 2];
      const v16bf b = __builtin_shufflevector(b0, b1,
          0, 1, 2, 3, 4, 5, 6, 7, 8, 9, 10, 11, 12, 13, 14, 15);
      acc[t] = __builtin_amdgcn_wmma_f32_16x16x32_bf16(
          /*neg_a=*/false, a, /*neg_b=*/false, b,
          /*c_mod=*/(short)0, acc[t], /*reuse_a=*/false, /*reuse_b=*/false);
    }
  }

  // C/D layout: VGPR v -> row m0 + v + 8*half, col = n0 + t*16 + l15
#pragma unroll
  for (int t = 0; t < NT; ++t) {
    const int col = n0 + t * 16 + l15;
    const float bb = bias ? bias[col] : 0.0f;
#pragma unroll
    for (int v = 0; v < 8; ++v) {
      const int row = m0 + v + half * 8;
      C[(size_t)row * Nc + col] = acc[t][v] + bb;
    }
  }
}

// ---------------------------------------------------------------------------
// f32 -> bf16 array conversion (layer-1 input staging)
// ---------------------------------------------------------------------------
__global__ void gat_cvt_bf16(const float* __restrict__ in,
                             __bf16* __restrict__ out, long long n)
{
  const long long i = (long long)blockIdx.x * blockDim.x + threadIdx.x;
  if (i < n) out[i] = to_bf16(in[i]);
}

// ---------------------------------------------------------------------------
// Weight transpose + f32 -> bf16:  W[K,Nc] -> Wt[Nc,K]
// ---------------------------------------------------------------------------
__global__ void gat_transpose_bf16(const float* __restrict__ W,
                                   __bf16* __restrict__ Wt, int K, int Nc)
{
  const int i = blockIdx.x * blockDim.x + threadIdx.x;
  if (i >= K * Nc) return;
  const int k = i / Nc;
  const int n = i - k * Nc;
  Wt[(size_t)n * K + k] = to_bf16(W[i]);
}

// ---------------------------------------------------------------------------
// alpha_src / alpha_dst:  [N, heads] dot-products over the hid channels
// ---------------------------------------------------------------------------
__global__ void gat_alpha(const float* __restrict__ h,
                          const float* __restrict__ a_src,
                          const float* __restrict__ a_dst,
                          float* __restrict__ as_, float* __restrict__ ad_,
                          int n, int heads, int hid)
{
  const int i = blockIdx.x * blockDim.x + threadIdx.x;
  if (i >= n * heads) return;
  const int node = i / heads;
  const int hh   = i - node * heads;
  const float* hp  = h + ((size_t)node * heads + hh) * hid;
  const float* avs = a_src + hh * hid;
  const float* avd = a_dst + hh * hid;
  float s = 0.f, d = 0.f;
#pragma unroll 8
  for (int c = 0; c < hid; ++c) { s = fmaf(hp[c], avs[c], s); d = fmaf(hp[c], avd[c], d); }
  as_[i] = s;
  ad_[i] = d;
}

// ---------------------------------------------------------------------------
// Edge pass 1: e = leaky_relu(as[src] + ad[dst]); segment max over dst
// ---------------------------------------------------------------------------
__global__ void gat_edge_score_max(const int* __restrict__ ei,
                                   const float* __restrict__ as_,
                                   const float* __restrict__ ad_,
                                   float* __restrict__ e_w,
                                   unsigned int* __restrict__ emax_ord,
                                   int heads)
{
  const long long i = (long long)blockIdx.x * blockDim.x + threadIdx.x;
  if (i >= (long long)GAT_ETOT * heads) return;
  const int e  = (int)(i / heads);
  const int hh = (int)(i - (long long)e * heads);
  int s, d;
  if (e < GAT_E) { s = ei[e]; d = ei[GAT_E + e]; } else { s = e - GAT_E; d = s; }
  float v = as_[(size_t)s * heads + hh] + ad_[(size_t)d * heads + hh];
  v = (v > 0.f) ? v : v * GAT_SLOPE;
  e_w[i] = v;
  atomicMax(&emax_ord[(size_t)d * heads + hh], f32_ord(v));
}

// ---------------------------------------------------------------------------
// Edge pass 2: w = exp(e - max[dst]); segment sum denom. Also prefetch the
// h[src] row (global_prefetch) to warm L2 for the scatter pass.
// ---------------------------------------------------------------------------
__global__ void gat_edge_exp_sum(const int* __restrict__ ei,
                                 const unsigned int* __restrict__ emax_ord,
                                 float* __restrict__ e_w,
                                 float* __restrict__ denom,
                                 const float* __restrict__ h, int F,
                                 int heads)
{
  const long long i = (long long)blockIdx.x * blockDim.x + threadIdx.x;
  if (i >= (long long)GAT_ETOT * heads) return;
  const int e  = (int)(i / heads);
  const int hh = (int)(i - (long long)e * heads);
  const int d  = (e < GAT_E) ? ei[GAT_E + e] : (e - GAT_E);
  const float m = ord_f32(emax_ord[(size_t)d * heads + hh]);
  const float w = __expf(e_w[i] - m);
  e_w[i] = w;
  atomicAdd(&denom[(size_t)d * heads + hh], w);
  if (hh == 0 && e < GAT_E) {
    // warm the random-gather line for edge pass 3 (h is L2-resident)
    __builtin_prefetch(h + (size_t)ei[e] * F, 0, 1);
  }
}

// ---------------------------------------------------------------------------
// Edge pass 3: out[dst] += h[src] * (w / denom[dst])   (per head, hid chans)
// ---------------------------------------------------------------------------
__global__ void gat_edge_scatter(const int* __restrict__ ei,
                                 const float* __restrict__ h,
                                 const float* __restrict__ e_w,
                                 const float* __restrict__ denom,
                                 float* __restrict__ out,
                                 int heads, int hid)
{
  const long long i = (long long)blockIdx.x * blockDim.x + threadIdx.x;
  if (i >= (long long)GAT_ETOT * heads) return;
  const int e  = (int)(i / heads);
  const int hh = (int)(i - (long long)e * heads);
  int s, d;
  if (e < GAT_E) { s = ei[e]; d = ei[GAT_E + e]; } else { s = e - GAT_E; d = s; }
  const float alpha = e_w[i] / (denom[(size_t)d * heads + hh] + 1e-16f);
  const float* hp = h   + ((size_t)s * heads + hh) * hid;
  float*       op = out + ((size_t)d * heads + hh) * hid;
#pragma unroll 8
  for (int c = 0; c < hid; ++c) atomicAdd(&op[c], hp[c] * alpha);
}

// ---------------------------------------------------------------------------
// bias + ELU; writes f32 result and bf16 mirror (next layer's WMMA A operand)
// ---------------------------------------------------------------------------
__global__ void gat_bias_elu(const float* __restrict__ in,
                             const float* __restrict__ bias,
                             float* __restrict__ out_f32,
                             __bf16* __restrict__ out_bf16,
                             long long n, int F)
{
  const long long i = (long long)blockIdx.x * blockDim.x + threadIdx.x;
  if (i >= n) return;
  const int c = (int)(i % F);
  float v = in[i] + bias[c];
  v = (v > 0.f) ? v : (__expf(v) - 1.0f);
  out_f32[i]  = v;
  out_bf16[i] = to_bf16(v);
}

// ---------------------------------------------------------------------------
// Fill / pool / divide
// ---------------------------------------------------------------------------
__global__ void gat_fill_u32(unsigned int* __restrict__ p, unsigned int v, long long n)
{
  long long i = (long long)blockIdx.x * blockDim.x + threadIdx.x;
  const long long stride = (long long)gridDim.x * blockDim.x;
  for (; i < n; i += stride) p[i] = v;
}

__global__ void gat_pool_scatter(const float* __restrict__ h,
                                 const int* __restrict__ batch,
                                 float* __restrict__ pooled,
                                 float* __restrict__ counts, int n, int F)
{
  const int i = blockIdx.x * blockDim.x + threadIdx.x;
  if (i >= n) return;
  const int g = batch[i];
  atomicAdd(&counts[g], 1.0f);
  const float* hp = h + (size_t)i * F;
  float* pp = pooled + (size_t)g * F;
#pragma unroll 8
  for (int c = 0; c < F; ++c) atomicAdd(&pp[c], hp[c]);
}

__global__ void gat_pool_div(float* __restrict__ pooled,
                             __bf16* __restrict__ pooled16,
                             const float* __restrict__ counts, int n, int F)
{
  const int i = blockIdx.x * blockDim.x + threadIdx.x;
  if (i >= n) return;
  const float v = pooled[i] / fmaxf(counts[i / F], 1.0f);
  pooled[i]   = v;
  pooled16[i] = to_bf16(v);
}

// ---------------------------------------------------------------------------
// Host-side helpers
// ---------------------------------------------------------------------------
static inline void fill_async(void* p, unsigned int v, long long n_u32, hipStream_t s) {
  int blocks = (int)((n_u32 + 255) / 256);
  if (blocks > 1024) blocks = 1024;
  if (blocks < 1) blocks = 1;
  gat_fill_u32<<<blocks, 256, 0, s>>>((unsigned int*)p, v, n_u32);
}

struct GatScratch {
  float*        bufH;    // [N,128] GEMM output h
  float*        bufA;    // [N,128] aggregated / next f32 input
  __bf16*       A16;     // [N,128] bf16 mirror of current layer input
  __bf16*       Wt16;    // [128,128] transposed bf16 weights (per layer reuse)
  float*        as_;
  float*        ad_;
  unsigned int* emax;
  float*        denom;
  float*        e_w;
  float*        pooled;
  __bf16*       pooled16;
  float*        counts;
};

template <int NT>
static void launch_gemm128(const __bf16* A16, const __bf16* Wt16,
                           const float* bias, float* C, int M, int Nc,
                           hipStream_t stream)
{
  const int tiles = (M >> 4) * (Nc / (16 * NT));
  gat_wmma_gemm<128, NT><<<(tiles + 7) / 8, 256, 0, stream>>>(A16, Wt16, bias, C, M, Nc);
}

static void run_gat_layer(const GatScratch& S,
                          const float* W, const float* a_s, const float* a_d,
                          const float* bias, int heads, int hid, int F_in,
                          const int* ei, hipStream_t stream)
{
  const int F_out = heads * hid;
  // 0) stage transposed bf16 weights
  gat_transpose_bf16<<<(F_in * F_out + 255) / 256, 256, 0, stream>>>(W, S.Wt16, F_in, F_out);
  // 1) h = A16 @ W  (WMMA)
  if (F_out == 128)
    launch_gemm128<4>(S.A16, S.Wt16, nullptr, S.bufH, GAT_N, F_out, stream);
  else
    launch_gemm128<2>(S.A16, S.Wt16, nullptr, S.bufH, GAT_N, F_out, stream);
  // 2) alpha_src / alpha_dst
  gat_alpha<<<(GAT_N * heads + 255) / 256, 256, 0, stream>>>(
      S.bufH, a_s, a_d, S.as_, S.ad_, GAT_N, heads, hid);
  // 3) reset accumulators (0 is ordered-encoding below any finite float)
  fill_async(S.emax,  0u, (long long)GAT_N * heads, stream);
  fill_async(S.denom, 0u, (long long)GAT_N * heads, stream);
  fill_async(S.bufA,  0u, (long long)GAT_N * F_out, stream);
  // 4) edge softmax + scatter
  const long long ne = (long long)GAT_ETOT * heads;
  const int eb = (int)((ne + 255) / 256);
  gat_edge_score_max<<<eb, 256, 0, stream>>>(ei, S.as_, S.ad_, S.e_w, S.emax, heads);
  gat_edge_exp_sum<<<eb, 256, 0, stream>>>(ei, S.emax, S.e_w, S.denom, S.bufH, F_out, heads);
  gat_edge_scatter<<<eb, 256, 0, stream>>>(ei, S.bufH, S.e_w, S.denom, S.bufA, heads, hid);
  // 5) bias + ELU (f32 + bf16 mirror for next layer's GEMM)
  const long long nel = (long long)GAT_N * F_out;
  gat_bias_elu<<<(int)((nel + 255) / 256), 256, 0, stream>>>(
      S.bufA, bias, S.bufA, S.A16, nel, F_out);
}

extern "C" void kernel_launch(void* const* d_in, const int* in_sizes, int n_in,
                              void* d_out, int out_size, void* d_ws, size_t ws_size,
                              hipStream_t stream)
{
  (void)in_sizes; (void)n_in; (void)out_size; (void)ws_size;
  const float* x     = (const float*)d_in[0];
  const int*   ei    = (const int*)  d_in[1];
  const int*   batch = (const int*)  d_in[2];
  const float* W1  = (const float*)d_in[3];
  const float* a1s = (const float*)d_in[4];
  const float* a1d = (const float*)d_in[5];
  const float* b1  = (const float*)d_in[6];
  const float* W2  = (const float*)d_in[7];
  const float* a2s = (const float*)d_in[8];
  const float* a2d = (const float*)d_in[9];
  const float* b2  = (const float*)d_in[10];
  const float* W3  = (const float*)d_in[11];
  const float* a3s = (const float*)d_in[12];
  const float* a3d = (const float*)d_in[13];
  const float* b3  = (const float*)d_in[14];
  const float* W4  = (const float*)d_in[15];
  const float* a4s = (const float*)d_in[16];
  const float* a4d = (const float*)d_in[17];
  const float* b4  = (const float*)d_in[18];
  const float* fcW = (const float*)d_in[19];
  const float* fcb = (const float*)d_in[20];

  // ---- carve workspace (256B aligned slices) ----
  char* ws = (char*)d_ws;
  auto carve = [&](size_t bytes) -> void* {
    void* p = (void*)ws;
    ws += (bytes + 255) & ~(size_t)255;
    return p;
  };
  GatScratch S;
  S.bufH     = (float*)       carve((size_t)GAT_N * 128 * 4);
  S.bufA     = (float*)       carve((size_t)GAT_N * 128 * 4);
  S.A16      = (__bf16*)      carve((size_t)GAT_N * 128 * 2);
  S.Wt16     = (__bf16*)      carve((size_t)128 * 128 * 2);
  S.as_      = (float*)       carve((size_t)GAT_N * GAT_HEADS * 4);
  S.ad_      = (float*)       carve((size_t)GAT_N * GAT_HEADS * 4);
  S.emax     = (unsigned int*)carve((size_t)GAT_N * GAT_HEADS * 4);
  S.denom    = (float*)       carve((size_t)GAT_N * GAT_HEADS * 4);
  S.e_w      = (float*)       carve((size_t)GAT_ETOT * GAT_HEADS * 4);
  S.pooled   = (float*)       carve((size_t)GAT_G * GAT_HID * 4);
  S.pooled16 = (__bf16*)      carve((size_t)GAT_G * GAT_HID * 2);
  S.counts   = (float*)       carve((size_t)GAT_G * 4);

  // ---- stage layer-1 input as bf16 ----
  gat_cvt_bf16<<<(GAT_N * GAT_IN + 255) / 256, 256, 0, stream>>>(
      x, S.A16, (long long)GAT_N * GAT_IN);

  // ---- 4 GAT layers ----
  run_gat_layer(S, W1, a1s, a1d, b1, GAT_HEADS, GAT_HID, GAT_IN,            ei, stream);
  run_gat_layer(S, W2, a2s, a2d, b2, GAT_HEADS, GAT_HID, GAT_HEADS*GAT_HID, ei, stream);
  run_gat_layer(S, W3, a3s, a3d, b3, GAT_HEADS, GAT_HID, GAT_HEADS*GAT_HID, ei, stream);
  run_gat_layer(S, W4, a4s, a4d, b4, 1,         GAT_HID, GAT_HEADS*GAT_HID, ei, stream);

  // ---- global mean pool over graphs ----
  fill_async(S.pooled, 0u, (long long)GAT_G * GAT_HID, stream);
  fill_async(S.counts, 0u, (long long)GAT_G, stream);
  gat_pool_scatter<<<(GAT_N + 255) / 256, 256, 0, stream>>>(
      S.bufA, batch, S.pooled, S.counts, GAT_N, GAT_HID);
  gat_pool_div<<<(GAT_G * GAT_HID + 255) / 256, 256, 0, stream>>>(
      S.pooled, S.pooled16, S.counts, GAT_G * GAT_HID, GAT_HID);

  // ---- fc: [G,HID] @ [HID,OUT] + fcb -> d_out  (WMMA, K=32) ----
  gat_transpose_bf16<<<(GAT_HID * GAT_OUT + 255) / 256, 256, 0, stream>>>(
      fcW, S.Wt16, GAT_HID, GAT_OUT);
  {
    const int tiles = (GAT_G >> 4) * (GAT_OUT / 64);   // NT=4 -> 8 tiles
    gat_wmma_gemm<32, 4><<<(tiles + 7) / 8, 256, 0, stream>>>(
        S.pooled16, S.Wt16, fcb, (float*)d_out, GAT_G, GAT_OUT);
  }
}
